// MultiHeadAttention_85633057948104
// MI455X (gfx1250) — matrix operand order repaired and checked by reading, not compiled, not run on hardware
//
#include <hip/hip_runtime.h>
#include <hip/hip_bf16.h>

typedef __attribute__((ext_vector_type(16))) _Float16 v16h;
typedef __attribute__((ext_vector_type(8)))  _Float16 v8h;
typedef __attribute__((ext_vector_type(8)))  float    v8f;

union AV { v16h v; v8h h[2]; };

#if __has_builtin(__builtin_amdgcn_tensor_load_to_lds) && __has_builtin(__builtin_amdgcn_s_wait_tensorcnt)
#define HAVE_TDM 1
#endif

// ---------------------------------------------------------------------------
// WMMA helper (V_WMMA_F32_16X16X32_F16, wave32)
// ---------------------------------------------------------------------------
__device__ __forceinline__ v8f wmma_f16(v16h a, v16h b, v8f c) {
    return __builtin_amdgcn_wmma_f32_16x16x32_f16(
        /*neg_a=*/false, a, /*neg_b=*/false, b,
        /*c_mod=*/(short)0, c, /*reuse_a=*/false, /*reuse_b=*/false);
}

// A-matrix (16x32 f16) load, row-major src with leading dim ld.
// lanes 0-15: row M=l16, K {0..7,16..23}; lanes 16-31: K {8..15,24..31}.
__device__ __forceinline__ v16h load_A16(const _Float16* __restrict__ src,
                                         long ld, int m0, int k0, int lane) {
    int l16 = lane & 15, hf = lane >> 4;
    const _Float16* p = src + (long)(m0 + l16) * ld + k0 + hf * 8;
    AV u;
    u.h[0] = *(const v8h*)(p);
    u.h[1] = *(const v8h*)(p + 16);
    return u.v;
}

// B-matrix (32x16 f16) load; element (k,n) at src[n*ld + k] (K-contiguous per column).
// lanes 0-15: col N=l16, K {0..7,16..23}; lanes 16-31: K {8..15,24..31}.
__device__ __forceinline__ v16h load_B16(const _Float16* __restrict__ src,
                                         long ld, int k0, int n0, int lane) {
    int l16 = lane & 15, hf = lane >> 4;
    const _Float16* p = src + (long)(n0 + l16) * ld + k0 + hf * 8;
    AV u;
    u.h[0] = *(const v8h*)(p);
    u.h[1] = *(const v8h*)(p + 16);
    return u.v;
}

// B-operand load from an LDS-resident row-major tile (ld = 64 halves).
__device__ __forceinline__ v16h load_B_lds(const _Float16* base, int n0, int c, int lane) {
    int l16 = lane & 15, hf = lane >> 4;
    const _Float16* p = base + (n0 + l16) * 64 + c + hf * 8;
    AV u;
    u.h[0] = *(const v8h*)(p);
    u.h[1] = *(const v8h*)(p + 16);
    return u.v;
}

// ---------------------------------------------------------------------------
// TDM: async DMA of a 64x64 f16 tile (row stride `stride_elems`) into LDS.
// D# built per CDNA5 ISA 8.3/8.4: group0 = count|lds_addr|global_addr|type=2,
// group1 = data_size=2B, tensor_dim0/1 = tile_dim0/1 = 64, dim0 stride.
// ---------------------------------------------------------------------------
#if defined(HAVE_TDM)
typedef __attribute__((ext_vector_type(4))) unsigned int u32x4;
typedef __attribute__((ext_vector_type(8))) int i32x8;
typedef __attribute__((ext_vector_type(4))) int i32x4;

__device__ __forceinline__ void tdm_load_tile64x64(const _Float16* gsrc, int stride_elems,
                                                   _Float16* lds_dst) {
    unsigned long long ga = (unsigned long long)gsrc;
    unsigned lds_off = (unsigned)(unsigned long long)lds_dst;   // low 32 bits = LDS byte offset
    u32x4 g0;
    g0[0] = 1u;                                                 // count=1 (valid descriptor)
    g0[1] = lds_off;                                            // lds_addr (bytes)
    g0[2] = (unsigned)(ga & 0xffffffffull);                     // global_addr[31:0]
    g0[3] = (unsigned)((ga >> 32) & 0x1ffffffull) | (2u << 30); // global_addr[56:32] | type=2
    i32x8 g1;
    g1[0] = 1 << 16;        // workgroup_mask=0 (not in cluster), data_size=1 (2 bytes)
    g1[1] = 64 << 16;       // tensor_dim0[15:0] = 64
    g1[2] = 64 << 16;       // tensor_dim0[31:16]=0 | tensor_dim1[15:0] = 64
    g1[3] = 64 << 16;       // tensor_dim1[31:16]=0 | tile_dim0 = 64
    g1[4] = 64;             // tile_dim1 = 64, tile_dim2 = 0 (2-D tile)
    g1[5] = stride_elems;   // tensor_dim0_stride[31:0] (data_size units)
    g1[6] = 0;              // stride hi | tensor_dim1_stride lo (unused for 2-D)
    g1[7] = 0;
    i32x4 z4 = {0, 0, 0, 0};
#if __clang_major__ >= 23
    i32x8 z8 = {0, 0, 0, 0, 0, 0, 0, 0};
    __builtin_amdgcn_tensor_load_to_lds(g0, g1, z4, z4, z8, 0);
#else
    __builtin_amdgcn_tensor_load_to_lds(g0, g1, z4, z4, 0);
#endif
}
#else
// Fallback: cooperative 128-thread copy of a 64x64 f16 tile into LDS.
__device__ __forceinline__ void stage_tile64x64(const _Float16* __restrict__ gsrc,
                                                int stride_elems, _Float16* lds_dst, int tid) {
    int row = tid >> 1, half = tid & 1;     // 2 threads per row, 32 halves each
    const _Float16* s = gsrc + (long)row * stride_elems + half * 32;
    _Float16* d = lds_dst + row * 64 + half * 32;
    *(v8h*)(d)      = *(const v8h*)(s);
    *(v8h*)(d + 8)  = *(const v8h*)(s + 8);
    *(v8h*)(d + 16) = *(const v8h*)(s + 16);
    *(v8h*)(d + 24) = *(const v8h*)(s + 24);
}
#endif

// ---------------------------------------------------------------------------
// Prep kernels
// ---------------------------------------------------------------------------
__global__ void cvt_f16_kernel(const float* __restrict__ in, _Float16* __restrict__ out, long n) {
    long i = (long)blockIdx.x * blockDim.x + threadIdx.x;
    if (i < n) out[i] = (_Float16)in[i];
}

__global__ void transpose_cvt_kernel(const float* __restrict__ in, _Float16* __restrict__ out,
                                     int K, int N) {
    long i = (long)blockIdx.x * blockDim.x + threadIdx.x;
    if (i < (long)K * N) {
        int k = (int)(i / N), n = (int)(i % N);
        out[(long)n * K + k] = (_Float16)in[i];
    }
}

// ---------------------------------------------------------------------------
// Kernel 1: QKV projection. One wave per 32(M) x 64(N) tile, K=1024 in steps
// of 32, software-pipelined. Epilogue adds bias and scatters f16 into
// Q,K [B,H,S,DH] and V^T [B,H,DH,S].
// ---------------------------------------------------------------------------
__global__ __launch_bounds__(32) void qkv_gemm_kernel(
    const _Float16* __restrict__ x16, const _Float16* __restrict__ wt,
    const float* __restrict__ bqkv,
    _Float16* __restrict__ Qb, _Float16* __restrict__ Kb, _Float16* __restrict__ Vtb) {
    const int lane = threadIdx.x & 31, l16 = lane & 15, hf = lane >> 4;
    const int TN = 3072 / 64;
    const int tm = blockIdx.x / TN, tn = blockIdx.x % TN;
    const int m0 = tm * 32, n0 = tn * 64;

    v8f acc[2][4];
    #pragma unroll
    for (int mh = 0; mh < 2; ++mh)
        #pragma unroll
        for (int s = 0; s < 4; ++s) acc[mh][s] = v8f{};

    v16h a0 = load_A16(x16, 1024, m0,      0, lane);
    v16h a1 = load_A16(x16, 1024, m0 + 16, 0, lane);
    v16h b0 = load_B16(wt, 1024, 0, n0 +  0, lane);
    v16h b1 = load_B16(wt, 1024, 0, n0 + 16, lane);
    v16h b2 = load_B16(wt, 1024, 0, n0 + 32, lane);
    v16h b3 = load_B16(wt, 1024, 0, n0 + 48, lane);

    for (int k0 = 0; k0 < 1024 - 32; k0 += 32) {
        const int kn = k0 + 32;
        v16h na0 = load_A16(x16, 1024, m0,      kn, lane);
        v16h na1 = load_A16(x16, 1024, m0 + 16, kn, lane);
        v16h nb0 = load_B16(wt, 1024, kn, n0 +  0, lane);
        v16h nb1 = load_B16(wt, 1024, kn, n0 + 16, lane);
        v16h nb2 = load_B16(wt, 1024, kn, n0 + 32, lane);
        v16h nb3 = load_B16(wt, 1024, kn, n0 + 48, lane);

        acc[0][0] = wmma_f16(a0, b0, acc[0][0]);
        acc[1][0] = wmma_f16(a1, b0, acc[1][0]);
        acc[0][1] = wmma_f16(a0, b1, acc[0][1]);
        acc[1][1] = wmma_f16(a1, b1, acc[1][1]);
        acc[0][2] = wmma_f16(a0, b2, acc[0][2]);
        acc[1][2] = wmma_f16(a1, b2, acc[1][2]);
        acc[0][3] = wmma_f16(a0, b3, acc[0][3]);
        acc[1][3] = wmma_f16(a1, b3, acc[1][3]);

        a0 = na0; a1 = na1; b0 = nb0; b1 = nb1; b2 = nb2; b3 = nb3;
    }
    acc[0][0] = wmma_f16(a0, b0, acc[0][0]);
    acc[1][0] = wmma_f16(a1, b0, acc[1][0]);
    acc[0][1] = wmma_f16(a0, b1, acc[0][1]);
    acc[1][1] = wmma_f16(a1, b1, acc[1][1]);
    acc[0][2] = wmma_f16(a0, b2, acc[0][2]);
    acc[1][2] = wmma_f16(a1, b2, acc[1][2]);
    acc[0][3] = wmma_f16(a0, b3, acc[0][3]);
    acc[1][3] = wmma_f16(a1, b3, acc[1][3]);

    const int sec = n0 >> 10;                       // 0=Q, 1=K, 2=V (wave-uniform)
    #pragma unroll
    for (int s = 0; s < 4; ++s) {
        const int n = n0 + s * 16 + l16;
        const float bias = bqkv[n];
        #pragma unroll
        for (int mh = 0; mh < 2; ++mh) {
            #pragma unroll
            for (int r = 0; r < 8; ++r) {
                const int m = m0 + mh * 16 + hf * 8 + r;
                const int srow = m & 2047, bb = m >> 11;
                _Float16 hv = (_Float16)(acc[mh][s][r] + bias);
                if (sec == 0) {
                    int h = n >> 6, dh = n & 63;
                    Qb[(((long)(bb * 16 + h)) * 2048 + srow) * 64 + dh] = hv;
                } else if (sec == 1) {
                    int n2 = n - 1024, h = n2 >> 6, dh = n2 & 63;
                    Kb[(((long)(bb * 16 + h)) * 2048 + srow) * 64 + dh] = hv;
                } else {
                    int n2 = n - 2048, h = n2 >> 6, dh = n2 & 63;
                    Vtb[(((long)(bb * 16 + h)) * 64 + dh) * 2048 + srow] = hv;
                }
            }
        }
    }
}

// ---------------------------------------------------------------------------
// Kernel 2: flash attention. Block = 128 threads (4 waves) per (b,h,64-row
// q-tile). K/V 64x64 tiles are DMA'd into LDS by the Tensor Data Mover
// (double-buffered: wave 0 issues tile kt+1, block computes tile kt, wave 0
// waits TENSORcnt before the end-of-iteration barrier). All 4 waves share the
// staged tiles -> 4x less L2 traffic than per-wave global loads.
// ---------------------------------------------------------------------------
__global__ __launch_bounds__(128) void attn_kernel(
    const _Float16* __restrict__ Qb, const _Float16* __restrict__ Kb,
    const _Float16* __restrict__ Vtb, _Float16* __restrict__ attn16) {
    __shared__ __align__(16) _Float16 lds_k[2][64 * 64];
    __shared__ __align__(16) _Float16 lds_v[2][64 * 64];
    __shared__ __align__(16) _Float16 p_lds[4 * 16 * 64];

    const int lane = threadIdx.x & 31, wave = threadIdx.x >> 5;
    const int l16 = lane & 15, hf = lane >> 4;
    const int qt = blockIdx.x & 31;              // S/64 = 32 q-tiles
    const int bh = blockIdx.x >> 5;              // b*16 + h
    const _Float16* Qh  = Qb  + (long)bh * 2048 * 64;
    const _Float16* Kh  = Kb  + (long)bh * 2048 * 64;
    const _Float16* Vth = Vtb + (long)bh * 64 * 2048;
    const int q0 = qt * 64 + wave * 16;

    const v16h aq0 = load_A16(Qh, 64, q0, 0, lane);
    const v16h aq1 = load_A16(Qh, 64, q0, 32, lane);

    float m_i[8], l_i[8];
    v8f o[4];
    #pragma unroll
    for (int r = 0; r < 8; ++r) { m_i[r] = -1e30f; l_i[r] = 0.f; }
    #pragma unroll
    for (int s = 0; s < 4; ++s) o[s] = v8f{};

    const float sc_log2 = 0.125f * 1.44269504088896f;  // 1/sqrt(64) * log2(e)

    // ---- prologue: stage tile 0 ----
#if defined(HAVE_TDM)
    if (wave == 0) {
        tdm_load_tile64x64(Kh, 64, &lds_k[0][0]);        // K tile: [key][dh], stride 64
        tdm_load_tile64x64(Vth, 2048, &lds_v[0][0]);     // V^T tile: [dh][key], stride 2048
        __builtin_amdgcn_s_wait_tensorcnt(0);
    }
#else
    stage_tile64x64(Kh, 64, &lds_k[0][0], threadIdx.x);
    stage_tile64x64(Vth, 2048, &lds_v[0][0], threadIdx.x);
#endif
    __syncthreads();

    for (int kt = 0; kt < 32; ++kt) {
        const int kbase = kt * 64;
        const int cur = kt & 1, nxt = cur ^ 1;

        // ---- issue DMA for next tile; completes while we compute this one ----
        if (kt + 1 < 32) {
#if defined(HAVE_TDM)
            if (wave == 0) {
                tdm_load_tile64x64(Kh + (long)(kbase + 64) * 64, 64, &lds_k[nxt][0]);
                tdm_load_tile64x64(Vth + (kbase + 64), 2048, &lds_v[nxt][0]);
            }
#else
            stage_tile64x64(Kh + (long)(kbase + 64) * 64, 64, &lds_k[nxt][0], threadIdx.x);
            stage_tile64x64(Vth + (kbase + 64), 2048, &lds_v[nxt][0], threadIdx.x);
#endif
        }

        // ---- scores: S = Q x K^T (8 WMMAs, B operands from LDS) ----
        v8f sc[4];
        #pragma unroll
        for (int s = 0; s < 4; ++s) {
            v8f c = v8f{};
            c = wmma_f16(aq0, load_B_lds(&lds_k[cur][0], s * 16, 0,  lane), c);
            c = wmma_f16(aq1, load_B_lds(&lds_k[cur][0], s * 16, 32, lane), c);
            sc[s] = c;
        }

        // ---- online softmax (row r = element r, spread over 16-lane group) ----
        #pragma unroll
        for (int s = 0; s < 4; ++s)
            #pragma unroll
            for (int r = 0; r < 8; ++r) sc[s][r] *= sc_log2;

        float mnew[8], alpha[8];
        #pragma unroll
        for (int r = 0; r < 8; ++r) {
            float mx = fmaxf(fmaxf(sc[0][r], sc[1][r]), fmaxf(sc[2][r], sc[3][r]));
            #pragma unroll
            for (int off = 1; off < 16; off <<= 1) mx = fmaxf(mx, __shfl_xor(mx, off, 32));
            mnew[r]  = fmaxf(m_i[r], mx);
            alpha[r] = exp2f(m_i[r] - mnew[r]);
        }
        #pragma unroll
        for (int s = 0; s < 4; ++s)
            #pragma unroll
            for (int r = 0; r < 8; ++r) sc[s][r] = exp2f(sc[s][r] - mnew[r]);
        #pragma unroll
        for (int r = 0; r < 8; ++r) {
            float rs = sc[0][r] + sc[1][r] + sc[2][r] + sc[3][r];
            #pragma unroll
            for (int off = 1; off < 16; off <<= 1) rs += __shfl_xor(rs, off, 32);
            l_i[r] = l_i[r] * alpha[r] + rs;
            m_i[r] = mnew[r];
        }
        #pragma unroll
        for (int s = 0; s < 4; ++s)
            #pragma unroll
            for (int r = 0; r < 8; ++r) o[s][r] *= alpha[r];

        // ---- stage P (f16) through LDS: C-layout -> A-layout ----
        #pragma unroll
        for (int s = 0; s < 4; ++s)
            #pragma unroll
            for (int r = 0; r < 8; ++r)
                p_lds[(wave * 16 + hf * 8 + r) * 64 + s * 16 + l16] = (_Float16)sc[s][r];
        __syncthreads();

        AV up0, up1;
        {
            const int rowbase = (wave * 16 + l16) * 64;
            up0.h[0] = *(const v8h*)&p_lds[rowbase + 0  + hf * 8];
            up0.h[1] = *(const v8h*)&p_lds[rowbase + 16 + hf * 8];
            up1.h[0] = *(const v8h*)&p_lds[rowbase + 32 + hf * 8];
            up1.h[1] = *(const v8h*)&p_lds[rowbase + 48 + hf * 8];
        }
        // ---- O += P x V (8 WMMAs, V from LDS) ----
        #pragma unroll
        for (int s = 0; s < 4; ++s) {
            o[s] = wmma_f16(up0.v, load_B_lds(&lds_v[cur][0], s * 16, 0,  lane), o[s]);
            o[s] = wmma_f16(up1.v, load_B_lds(&lds_v[cur][0], s * 16, 32, lane), o[s]);
        }

#if defined(HAVE_TDM)
        if (wave == 0) __builtin_amdgcn_s_wait_tensorcnt(0);
#endif
        __syncthreads();   // next-tile DMA complete + p_lds/K/V reads done
    }

    // ---- finalize: O /= l, write attn f16 in [B*S, D] ----
    const int b = bh >> 4, h = bh & 15;
    #pragma unroll
    for (int r = 0; r < 8; ++r) {
        const float inv = 1.0f / l_i[r];
        const int srow = qt * 64 + wave * 16 + hf * 8 + r;
        #pragma unroll
        for (int s = 0; s < 4; ++s) {
            const int col = h * 64 + s * 16 + l16;
            attn16[((long)b * 2048 + srow) * 1024 + col] = (_Float16)(o[s][r] * inv);
        }
    }
}

// ---------------------------------------------------------------------------
// Kernel 3: output projection, 32x64 tile per wave, pipelined like kernel 1,
// f32 result + bias directly to d_out.
// ---------------------------------------------------------------------------
__global__ __launch_bounds__(32) void out_gemm_kernel(
    const _Float16* __restrict__ a16, const _Float16* __restrict__ wt,
    const float* __restrict__ bout, float* __restrict__ out) {
    const int lane = threadIdx.x & 31, l16 = lane & 15, hf = lane >> 4;
    const int TN = 1024 / 64;
    const int tm = blockIdx.x / TN, tn = blockIdx.x % TN;
    const int m0 = tm * 32, n0 = tn * 64;

    v8f acc[2][4];
    #pragma unroll
    for (int mh = 0; mh < 2; ++mh)
        #pragma unroll
        for (int s = 0; s < 4; ++s) acc[mh][s] = v8f{};

    v16h a0 = load_A16(a16, 1024, m0,      0, lane);
    v16h a1 = load_A16(a16, 1024, m0 + 16, 0, lane);
    v16h b0 = load_B16(wt, 1024, 0, n0 +  0, lane);
    v16h b1 = load_B16(wt, 1024, 0, n0 + 16, lane);
    v16h b2 = load_B16(wt, 1024, 0, n0 + 32, lane);
    v16h b3 = load_B16(wt, 1024, 0, n0 + 48, lane);

    for (int k0 = 0; k0 < 1024 - 32; k0 += 32) {
        const int kn = k0 + 32;
        v16h na0 = load_A16(a16, 1024, m0,      kn, lane);
        v16h na1 = load_A16(a16, 1024, m0 + 16, kn, lane);
        v16h nb0 = load_B16(wt, 1024, kn, n0 +  0, lane);
        v16h nb1 = load_B16(wt, 1024, kn, n0 + 16, lane);
        v16h nb2 = load_B16(wt, 1024, kn, n0 + 32, lane);
        v16h nb3 = load_B16(wt, 1024, kn, n0 + 48, lane);

        acc[0][0] = wmma_f16(a0, b0, acc[0][0]);
        acc[1][0] = wmma_f16(a1, b0, acc[1][0]);
        acc[0][1] = wmma_f16(a0, b1, acc[0][1]);
        acc[1][1] = wmma_f16(a1, b1, acc[1][1]);
        acc[0][2] = wmma_f16(a0, b2, acc[0][2]);
        acc[1][2] = wmma_f16(a1, b2, acc[1][2]);
        acc[0][3] = wmma_f16(a0, b3, acc[0][3]);
        acc[1][3] = wmma_f16(a1, b3, acc[1][3]);

        a0 = na0; a1 = na1; b0 = nb0; b1 = nb1; b2 = nb2; b3 = nb3;
    }
    acc[0][0] = wmma_f16(a0, b0, acc[0][0]);
    acc[1][0] = wmma_f16(a1, b0, acc[1][0]);
    acc[0][1] = wmma_f16(a0, b1, acc[0][1]);
    acc[1][1] = wmma_f16(a1, b1, acc[1][1]);
    acc[0][2] = wmma_f16(a0, b2, acc[0][2]);
    acc[1][2] = wmma_f16(a1, b2, acc[1][2]);
    acc[0][3] = wmma_f16(a0, b3, acc[0][3]);
    acc[1][3] = wmma_f16(a1, b3, acc[1][3]);

    #pragma unroll
    for (int s = 0; s < 4; ++s) {
        const int n = n0 + s * 16 + l16;
        const float bias = bout[n];
        #pragma unroll
        for (int mh = 0; mh < 2; ++mh) {
            #pragma unroll
            for (int r = 0; r < 8; ++r) {
                const int m = m0 + mh * 16 + hf * 8 + r;
                out[(long)m * 1024 + n] = acc[mh][s][r] + bias;
            }
        }
    }
}

// ---------------------------------------------------------------------------
// Launch
// ---------------------------------------------------------------------------
extern "C" void kernel_launch(void* const* d_in, const int* in_sizes, int n_in,
                              void* d_out, int out_size, void* d_ws, size_t ws_size,
                              hipStream_t stream) {
    (void)in_sizes; (void)n_in; (void)out_size; (void)ws_size;
    const float* x    = (const float*)d_in[0];
    const float* Wqkv = (const float*)d_in[1];
    const float* bqkv = (const float*)d_in[2];
    const float* Wout = (const float*)d_in[3];
    const float* bout = (const float*)d_in[4];
    float* out = (float*)d_out;

    // Workspace layout (f16 elements), total 48 MB.
    _Float16* ws     = (_Float16*)d_ws;
    _Float16* x16    = ws;                                  // 4096*1024
    _Float16* wqkvt  = x16   + (size_t)4096 * 1024;         // 3072*1024 (transposed)
    _Float16* woutt  = wqkvt + (size_t)3072 * 1024;         // 1024*1024 (transposed)
    _Float16* Qb     = woutt + (size_t)1024 * 1024;         // [B,H,S,DH] = 4M
    _Float16* Kb     = Qb    + (size_t)4194304;
    _Float16* Vtb    = Kb    + (size_t)4194304;             // [B,H,DH,S]
    _Float16* attn16 = Vtb   + (size_t)4194304;             // [B*S, D]

    cvt_f16_kernel<<<16384, 256, 0, stream>>>(x, x16, (long)4096 * 1024);
    transpose_cvt_kernel<<<12288, 256, 0, stream>>>(Wqkv, wqkvt, 1024, 3072);
    transpose_cvt_kernel<<<4096, 256, 0, stream>>>(Wout, woutt, 1024, 1024);

    qkv_gemm_kernel<<<128 * 48, 32, 0, stream>>>(x16, wqkvt, bqkv, Qb, Kb, Vtb);
    attn_kernel<<<1024, 128, 0, stream>>>(Qb, Kb, Vtb, attn16);
    out_gemm_kernel<<<128 * 16, 32, 0, stream>>>(attn16, woutt, bout, out);
}
